// TemporalContrastiveLoss_15109694947825
// MI455X (gfx1250) — matrix operand order repaired and checked by reading, not compiled, not run on hardware
//
#include <hip/hip_runtime.h>

// CDNA5 / gfx1250: wave32, WMMA bf16 16x16x32.
typedef __attribute__((ext_vector_type(16))) __bf16 v16bf;
typedef __attribute__((ext_vector_type(8)))  float  v8f;

#define WMMA_BF16(a, b, c) \
  __builtin_amdgcn_wmma_f32_16x16x32_bf16(false, (a), false, (b), (short)0, (c), false, false)

__device__ __forceinline__ unsigned short f32_to_bf16_rne(float f) {
  unsigned u = __float_as_uint(f);
  unsigned r = (u + 0x7FFFu + ((u >> 16) & 1u)) >> 16;
  return (unsigned short)r;
}

// ---------------------------------------------------------------------------
// Phase 1: L2-normalize rows; emit fp32 copy + bf16 hi/lo fragments pre-packed
// in the ISA WMMA VGPR layouts (A-layout and B-layout).
//   A 16-bit 16x32 (ISA 7.12.2): lane = h*16 + (row&15);
//     element e -> K = (e&7) + (e>=8 ? 16 : 0) + 8*h   (inverse used here)
//   B 32x16: lane = h*16 + (row&15); element e -> K = (e&7) + (e>=8 ? 16 : 0) + 8*h (same as A)
// One wave per row (128 elems -> 4 per lane).
// ---------------------------------------------------------------------------
__global__ __launch_bounds__(256) void tcl_normalize_split(
    const float* __restrict__ sig, float* __restrict__ xnorm,
    unsigned short* __restrict__ hiA, unsigned short* __restrict__ loA,
    unsigned short* __restrict__ hiB, unsigned short* __restrict__ loB) {
  const int wave = threadIdx.x >> 5;
  const int lane = threadIdx.x & 31;
  const int row  = blockIdx.x * 8 + wave;

  float4 v = ((const float4*)(sig + (size_t)row * 128))[lane];
  float ss = v.x * v.x + v.y * v.y + v.z * v.z + v.w * v.w;
  ss += __shfl_xor(ss, 16, 32);
  ss += __shfl_xor(ss, 8, 32);
  ss += __shfl_xor(ss, 4, 32);
  ss += __shfl_xor(ss, 2, 32);
  ss += __shfl_xor(ss, 1, 32);
  const float scale = 1.0f / fmaxf(sqrtf(ss), 1e-12f);

  float xv[4] = {v.x * scale, v.y * scale, v.z * scale, v.w * scale};
  ((float4*)(xnorm + (size_t)row * 128))[lane] =
      make_float4(xv[0], xv[1], xv[2], xv[3]);

  const int m = row >> 4, rr = row & 15;
#pragma unroll
  for (int q = 0; q < 4; ++q) {
    const int k = lane * 4 + q;
    const float xf = xv[q];
    const unsigned short hi = f32_to_bf16_rne(xf);
    const float hif = __uint_as_float(((unsigned)hi) << 16);
    const unsigned short lo = f32_to_bf16_rne(xf - hif);
    const int kc = k >> 5, kk = k & 31;
    // A-layout inverse
    const int hA = (kk >> 3) & 1;
    const int eA = (kk & 7) + ((kk >= 16) ? 8 : 0);
    const size_t ia = ((size_t)(m * 4 + kc) * 32 + hA * 16 + rr) * 16 + eA;
    hiA[ia] = hi;
    loA[ia] = lo;
    // B-layout inverse
    const int hB = (kk >> 3) & 1;
    const int eB = (kk & 7) + ((kk >= 16) ? 8 : 0);
    const size_t ib = ((size_t)(m * 4 + kc) * 32 + hB * 16 + rr) * 16 + eB;
    hiB[ib] = hi;
    loB[ib] = lo;
  }
}

// ---------------------------------------------------------------------------
// Phase 2: band positives. pos_sum[i] = sum exp(10*cos(i,j)) over
// 0 < |idx_i - idx_j| <= 10 within the positional band (indices are arange).
// One wave per row; fp32 dot + wave butterfly reduce.
// ---------------------------------------------------------------------------
__global__ __launch_bounds__(128) void tcl_band(
    const float* __restrict__ xnorm, const int* __restrict__ indices,
    float* __restrict__ possum, float* __restrict__ poscnt, int B) {
  const int wave = threadIdx.x >> 5;
  const int lane = threadIdx.x & 31;
  const int i = blockIdx.x * 4 + wave;
  if (i >= B) return;

  const float4 xi = ((const float4*)(xnorm + (size_t)i * 128))[lane];
  const int idxi = indices[i];
  float pos = 0.f, cnt = 0.f;
  for (int j = i - 10; j <= i + 10; ++j) {
    if (j < 0 || j >= B || j == i) continue;   // wave-uniform
    int d = indices[j] - idxi;
    d = d < 0 ? -d : d;
    if (d == 0 || d > 10) continue;            // wave-uniform
    const float4 xj = ((const float4*)(xnorm + (size_t)j * 128))[lane];
    float dp = xi.x * xj.x + xi.y * xj.y + xi.z * xj.z + xi.w * xj.w;
    dp += __shfl_xor(dp, 16, 32);
    dp += __shfl_xor(dp, 8, 32);
    dp += __shfl_xor(dp, 4, 32);
    dp += __shfl_xor(dp, 2, 32);
    dp += __shfl_xor(dp, 1, 32);
    pos += __expf(dp * 10.0f);
    cnt += 1.0f;
  }
  if (lane == 0) {
    possum[i] = pos;
    poscnt[i] = cnt;
  }
}

// ---------------------------------------------------------------------------
// Phase 3: full GEMM (bf16x3 split) + exp + off-diagonal row sums.
// WG = 256 threads = 8 waves; wave w owns M-tile blockIdx.x*8+w.
// grid.y = 8 N-chunks. B-fragments are software-pipelined: prefetch tile nt+1
// from L2 into registers while computing tile nt from double-buffered LDS
// (one barrier per iteration). A-fragments (hi+lo, 4 K-chunks) in registers.
// Three independent WMMA accumulator chains (hi*hi, hi*lo, lo*hi) keep the
// matrix pipe busy (dep chain 4 instead of 12).
// Deterministic: each (m-tile, n-chunk) writes a private partial slot.
// ---------------------------------------------------------------------------
__global__ __launch_bounds__(256) void tcl_gemm_rowsum(
    const unsigned short* __restrict__ hiA, const unsigned short* __restrict__ loA,
    const unsigned short* __restrict__ hiB, const unsigned short* __restrict__ loB,
    float* __restrict__ parts, int B) {
  __shared__ __align__(32) unsigned short ldsB[2][2][4][32][16];  // [buf][hi/lo][kc][lane][e] = 16 KB

  const int tid  = threadIdx.x;
  const int wave = tid >> 5;
  const int lane = tid & 31;
  const int mTile = blockIdx.x * 8 + wave;
  const int nChunk = blockIdx.y;
  const int nTilesPerChunk = (B / 16) / 8;
  const int r = lane & 15, h = lane >> 4;

  // Hoist A fragments (invariant over the N loop).
  v16bf aHi[4], aLo[4];
#pragma unroll
  for (int kc = 0; kc < 4; ++kc) {
    const size_t off = ((size_t)(mTile * 4 + kc) * 32 + lane) * 16;
    aHi[kc] = *(const v16bf*)(hiA + off);
    aLo[kc] = *(const v16bf*)(loA + off);
  }

  float rsum[8];
#pragma unroll
  for (int v = 0; v < 8; ++v) rsum[v] = 0.f;

  // LDS staging role for this thread (fixed over loop): 256 threads move the
  // 8 KB B-panel (hi+lo, 4 kc) as one 32B chunk each.
  const int sel = tid >> 7;       // 0 = hi, 1 = lo
  const int rem = tid & 127;
  const int skc = rem >> 5;
  const int sln = rem & 31;
  const unsigned short* const sBase = sel ? loB : hiB;
  const size_t sStride16 = ((size_t)(nChunk * nTilesPerChunk) * 4 + skc) * 32 + sln;

  // Prefetch tile 0 into registers.
  uint4 g0, g1;
  {
    const uint4* s4 = (const uint4*)(sBase + sStride16 * 16);
    g0 = s4[0];
    g1 = s4[1];
  }

  for (int nt = 0; nt < nTilesPerChunk; ++nt) {
    const int buf = nt & 1;
    // Stage prefetched panel into this iteration's LDS buffer.
    {
      uint4* dst = (uint4*)&ldsB[buf][sel][skc][sln][0];
      dst[0] = g0;
      dst[1] = g1;
    }
    // Prefetch next tile (latency hidden behind this tile's compute).
    if (nt + 1 < nTilesPerChunk) {
      const uint4* s4 = (const uint4*)(sBase + (sStride16 + (size_t)(nt + 1) * 128) * 16);
      g0 = s4[0];
      g1 = s4[1];
    }
    __syncthreads();  // panel visible to all waves; prior-iter readers of the
                      // *other* buffer were already fenced by last barrier.

    v8f accHH = {}, accHL = {}, accLH = {};
#pragma unroll
    for (int kc = 0; kc < 4; ++kc) {
      const v16bf bHi = *(const v16bf*)&ldsB[buf][0][kc][lane][0];
      const v16bf bLo = *(const v16bf*)&ldsB[buf][1][kc][lane][0];
      accHH = WMMA_BF16(aHi[kc], bHi, accHH);
      accHL = WMMA_BF16(aHi[kc], bLo, accHL);
      accLH = WMMA_BF16(aLo[kc], bHi, accLH);
    }

    // D layout (16x16 f32): lane (r,h), VGPR v -> element (M = v + 8h, N = r).
    const int nTile = nChunk * nTilesPerChunk + nt;
    const int colg = nTile * 16 + r;
#pragma unroll
    for (int v = 0; v < 8; ++v) {
      const int rowg = mTile * 16 + h * 8 + v;
      const float sim = accHH[v] + (accHL[v] + accLH[v]);
      const float e = __expf(sim * 10.0f);     // sim / T, T = 0.1
      rsum[v] += (rowg == colg) ? 0.0f : e;    // exclude diagonal exactly
    }
  }

  // Reduce across the 16 lanes of each half (they hold the 16 N columns).
#pragma unroll
  for (int v = 0; v < 8; ++v) {
    float s = rsum[v];
    s += __shfl_xor(s, 1, 16);
    s += __shfl_xor(s, 2, 16);
    s += __shfl_xor(s, 4, 16);
    s += __shfl_xor(s, 8, 16);
    rsum[v] = s;
  }
  if (r == 0) {
#pragma unroll
    for (int v = 0; v < 8; ++v)
      parts[(size_t)nChunk * B + mTile * 16 + h * 8 + v] = rsum[v];
  }
}

// ---------------------------------------------------------------------------
// Phase 4: per-row loss and mean. Fixed-order sums -> deterministic.
// ---------------------------------------------------------------------------
__global__ __launch_bounds__(256) void tcl_finalize(
    const float* __restrict__ parts, const float* __restrict__ possum,
    const float* __restrict__ poscnt, float* __restrict__ out, int B) {
  __shared__ float sL[256];
  __shared__ float sN[256];
  const int tid = threadIdx.x;
  float accL = 0.f, accN = 0.f;
  for (int i = tid; i < B; i += 256) {
    float tot = 0.f;
#pragma unroll
    for (int p = 0; p < 8; ++p) tot += parts[(size_t)p * B + i];
    const float pos = possum[i];
    const float pc = poscnt[i];
    const float nc = (float)(B - 1) - pc;
    const bool valid = (pc > 0.f) && (nc > 0.f);
    const float per = -logf(pos / (tot + 1e-8f));  // tot = pos + neg (off-diag)
    accL += valid ? per : 0.f;
    accN += valid ? 1.f : 0.f;
  }
  sL[tid] = accL;
  sN[tid] = accN;
  __syncthreads();
  for (int s = 128; s > 0; s >>= 1) {
    if (tid < s) {
      sL[tid] += sL[tid + s];
      sN[tid] += sN[tid + s];
    }
    __syncthreads();
  }
  if (tid == 0) out[0] = (sN[0] > 0.f) ? (sL[0] / sN[0]) : 0.f;
}

// ---------------------------------------------------------------------------
extern "C" void kernel_launch(void* const* d_in, const int* in_sizes, int n_in,
                              void* d_out, int out_size, void* d_ws, size_t ws_size,
                              hipStream_t stream) {
  const float* sig = (const float*)d_in[0];
  const int* idx = (const int*)d_in[1];
  float* out = (float*)d_out;

  const int B = in_sizes[1];  // 8192 ; D fixed at 128 by the tiling below

  char* ws = (char*)d_ws;
  float* xnorm           = (float*)(ws);                       // B*128*4   = 4 MB
  unsigned short* hiA    = (unsigned short*)(ws + (4u << 20)); // B*128*2   = 2 MB
  unsigned short* loA    = (unsigned short*)(ws + (6u << 20));
  unsigned short* hiB    = (unsigned short*)(ws + (8u << 20));
  unsigned short* loB    = (unsigned short*)(ws + (10u << 20));
  float* possum          = (float*)(ws + (12u << 20));                 // 32 KB
  float* poscnt          = (float*)(ws + (12u << 20) + (1u << 15));   // 32 KB
  float* parts           = (float*)(ws + (12u << 20) + (1u << 16));   // 8*B*4 = 256 KB

  tcl_normalize_split<<<B / 8, 256, 0, stream>>>(sig, xnorm, hiA, loA, hiB, loB);
  tcl_band<<<B / 4, 128, 0, stream>>>(xnorm, idx, possum, poscnt, B);
  tcl_gemm_rowsum<<<dim3(B / 128, 8), 256, 0, stream>>>(hiA, loA, hiB, loB, parts, B);
  tcl_finalize<<<1, 256, 0, stream>>>(parts, possum, poscnt, out, B);
}
